// GNNAngle_21122649162275
// MI455X (gfx1250) — matrix operand order repaired and checked by reading, not compiled, not run on hardware
//
#include <hip/hip_runtime.h>
#include <math.h>

// ---- CDNA5 WMMA vector types -----------------------------------------------
typedef __attribute__((ext_vector_type(16))) _Float16 v16h;
typedef __attribute__((ext_vector_type(8)))  _Float16 v8h;
typedef __attribute__((ext_vector_type(8)))  float    v8f;

#define KNN        16
#define DIM        32
#define HID        128
#define LW         136   // padded LDS row stride (halves): bank-conflict break, 16B aligned
#define WAVES      8     // waves per block

// ---- LDS layout (dynamic shared, byte offsets) ------------------------------
#define SZ_WT      (HID * LW * 2)              // one transposed f16 weight matrix: 34816 B
#define OFF_W1     0
#define OFF_W2     (OFF_W1 + SZ_WT)
#define OFF_W3     (OFF_W2 + SZ_WT)            // ends at 104448
#define OFF_W4     (OFF_W3 + SZ_WT)            // float[128]
#define OFF_B1     (OFF_W4 + 512)
#define OFF_B2     (OFF_B1 + 512)
#define OFF_B3     (OFF_B2 + 512)
#define OFF_B4     (OFF_B3 + 512)              // 106496..106500
#define OFF_TAB    106504                      // 120 x u8 packed (i<<4)|j pair table
#define OFF_WAVE   106752                      // per-wave scratch base (256B aligned)
#define WAVE_STRIDE 21760                      // vh 16384 + bufA 4352 + gram 1024
#define SMEM_TOTAL (OFF_WAVE + WAVES * WAVE_STRIDE)  // 280832 B (< 320KB WGP LDS)

__device__ __forceinline__ v16h cat16(v8h lo, v8h hi) {
  return __builtin_shufflevector(lo, hi, 0,1,2,3,4,5,6,7,8,9,10,11,12,13,14,15);
}

__device__ __forceinline__ v8f wmma_f16(v16h a, v16h b, v8f c) {
  // D(16x16 f32) = A(16x32 f16) * B(32x16 f16) + C
  return __builtin_amdgcn_wmma_f32_16x16x32_f16(
      /*neg_a=*/false, a, /*neg_b=*/false, b,
      /*c_mod=*/(short)0, c, /*reuse_a=*/false, /*reuse_b=*/false);
}

// One 16xHID = 16xHID @ HIDxHID layer. A-chunks hoisted into registers once;
// B from transposed f16 weights wT[n*LW + k]; fused bias + tanh.
__device__ __forceinline__ void mlp_layer(const _Float16* in, const _Float16* wT,
                                          const float* bias, _Float16* out,
                                          int m, int hi) {
  v16h a[4];
#pragma unroll
  for (int ks = 0; ks < 4; ++ks) {
    // A operand, lane = row m: halves 0..7 -> K=k0+hi*8.., halves 8..15 -> +16
    const _Float16* ra = in + m * LW + ks * 32 + hi * 8;
    a[ks] = cat16(*(const v8h*)ra, *(const v8h*)(ra + 16));
  }
  for (int nt = 0; nt < 8; ++nt) {
    const int n0 = nt * 16;
    v8f c = {};
#pragma unroll
    for (int ks = 0; ks < 4; ++ks) {
      // B operand, lane = col n0+m: halves 0..7 -> K=k0+hi*8.., halves 8..15 -> +16
      const _Float16* rb = wT + (n0 + m) * LW + ks * 32 + hi * 8;
      v16h b = cat16(*(const v8h*)rb, *(const v8h*)(rb + 16));
      c = wmma_f16(a[ks], b, c);
    }
    const float bb = bias[n0 + m];
#pragma unroll
    for (int r = 0; r < 8; ++r)  // D: lane holds rows r+hi*8, col n0+m
      out[(r + hi * 8) * LW + n0 + m] = (_Float16)tanhf(c[r] + bb);
  }
}

__global__ void __launch_bounds__(32 * WAVES)
gnn_angle_wmma_kernel(const float* __restrict__ edge_attr,
                      const float* __restrict__ W1, const float* __restrict__ b1,
                      const float* __restrict__ W2, const float* __restrict__ b2,
                      const float* __restrict__ W3, const float* __restrict__ b3,
                      const float* __restrict__ W4, const float* __restrict__ b4,
                      float* __restrict__ out, int nNodes) {
  extern __shared__ char smem[];

  const int tid  = threadIdx.x;
  const int lane = tid & 31;
  const int wv   = tid >> 5;
  const int hi   = lane >> 4;  // 0: lanes 0-15, 1: lanes 16-31
  const int m    = lane & 15;  // row index for A/D, col index for B/D

  _Float16*      w1T = (_Float16*)(smem + OFF_W1);
  _Float16*      w2T = (_Float16*)(smem + OFF_W2);
  _Float16*      w3T = (_Float16*)(smem + OFF_W3);
  float*         w4f = (float*)(smem + OFF_W4);
  float*         b1f = (float*)(smem + OFF_B1);
  float*         b2f = (float*)(smem + OFF_B2);
  float*         b3f = (float*)(smem + OFF_B3);
  float*         b4f = (float*)(smem + OFF_B4);
  unsigned char* tab = (unsigned char*)(smem + OFF_TAB);

  char*     wbase = smem + OFF_WAVE + wv * WAVE_STRIDE;
  _Float16* vh    = (_Float16*)(wbase);           // [16 nodes][16 edges][32] f16
  _Float16* bufB  = (_Float16*)(wbase);           // overlays vh once Gram phase is done
  _Float16* bufA  = (_Float16*)(wbase + 16384);   // [16][LW] f16 features / activations
  float*    gram  = (float*)(wbase + 20736);      // [16][16] f32, wave-private

  // ---- stage weights to LDS: f16, transposed, W1 K-padded 120 -> 128 -------
  for (int idx = tid; idx < HID * HID; idx += 32 * WAVES) {
    const int n = idx >> 7, kk = idx & 127;
    w1T[n * LW + kk] = (_Float16)((kk < 120) ? W1[kk * HID + n] : 0.0f);
    w2T[n * LW + kk] = (_Float16)W2[kk * HID + n];
    w3T[n * LW + kk] = (_Float16)W3[kk * HID + n];
  }
  if (tid < HID) {
    w4f[tid] = W4[tid];
    b1f[tid] = b1[tid]; b2f[tid] = b2[tid]; b3f[tid] = b3[tid];
  }
  if (tid == 0) b4f[0] = b4[0];
  if (tid < 120) {  // upper-tri pair unranking table, packed nibbles
    int pp = tid, i = 0;
    while (pp >= 15 - i) { pp -= 15 - i; ++i; }
    tab[tid] = (unsigned char)((i << 4) | (i + 1 + pp));
  }

  int nodeBase = (blockIdx.x * WAVES + wv) * 16;
  if (nodeBase + 16 > nNodes) nodeBase = nNodes - 16;  // uniform-per-wave clamp

  // ---- load + L2-normalize 16 nodes x 16 edge vectors -> f16 in LDS --------
  for (int rr = lane; rr < 256; rr += 32) {
    const int nd = rr >> 4, ed = rr & 15;
    const float* src = edge_attr + ((size_t)(nodeBase + nd) * KNN + ed) * DIM;
    float buf[DIM];
    float ss = 0.f;
#pragma unroll
    for (int q = 0; q < DIM; q += 4) {
      float4 f = *(const float4*)(src + q);
      buf[q] = f.x; buf[q + 1] = f.y; buf[q + 2] = f.z; buf[q + 3] = f.w;
      ss += f.x * f.x + f.y * f.y + f.z * f.z + f.w * f.w;
    }
    const float inv = 1.0f / (sqrtf(ss) + 1e-12f);
    _Float16* dst = vh + nd * 512 + ed * 32;
#pragma unroll
    for (int q = 0; q < DIM; ++q) dst[q] = (_Float16)(buf[q] * inv);
  }

  // zero feature padding columns 120..127
  for (int rr = lane; rr < 128; rr += 32) {
    const int nd = rr >> 3, q = rr & 7;
    bufA[nd * LW + 120 + q] = (_Float16)0.f;
  }
  __syncthreads();  // vh + weights + tab visible to all lanes

  // per-lane gram-read / feature-write offsets, hoisted out of the node loop.
  // lanes with p>119 clamp to pair 119 (duplicate store of identical value).
  int goff[4], foff[4];
#pragma unroll
  for (int t = 0; t < 4; ++t) {
    int p = lane + t * 32;
    if (p > 119) p = 119;
    const int ij = tab[p];
    goff[t] = (ij >> 4) * 16 + (ij & 15);
    foff[t] = p;
  }

  // ---- per node: Gram = V*V^T via one 16x16x32 WMMA; angles via a 1KB
  //      wave-private LDS bounce with wave-local fencing (no s_barrier) ------
  for (int nd = 0; nd < 16; ++nd) {
    const _Float16* base = vh + nd * 512;
    // A = V (16x32): lane m row, chunks K=(0..7)+hi*8 and K=(16..23)+hi*8
    const _Float16* ra = base + m * 32 + hi * 8;
    v16h a = cat16(*(const v8h*)ra, *(const v8h*)(ra + 16));
    // B = V^T (32x16): col n=m is row m of V; chunks K=(0..7)+hi*8 and K=(16..23)+hi*8
    const _Float16* rb = base + m * 32 + hi * 8;
    v16h b = cat16(*(const v8h*)rb, *(const v8h*)(rb + 16));
    v8f c = {};
    c = wmma_f16(a, b, c);

#pragma unroll
    for (int r = 0; r < 8; ++r) gram[(r + hi * 8) * 16 + m] = c[r];
    // wave-local visibility: drains DScnt, no block barrier. LDS ops from a
    // wave are processed in order, so WAR on the next iteration is safe too.
    __builtin_amdgcn_fence(__ATOMIC_ACQ_REL, "workgroup");

    _Float16* frow = bufA + nd * LW;
#pragma unroll
    for (int t = 0; t < 4; ++t) {  // 120 acos across 32 lanes, unpredicated
      float g = gram[goff[t]];
      g = fminf(fmaxf(g, -1.0f + 1e-7f), 1.0f - 1e-7f);
      frow[foff[t]] = (_Float16)acosf(g);
    }
    __builtin_amdgcn_fence(__ATOMIC_ACQ_REL, "workgroup");
  }
  __syncthreads();  // features complete before GEMM reads them

  // ---- MLP: 3 x (16x128 @ 128x128) on WMMA, fused bias+tanh ----------------
  mlp_layer(bufA, w1T, b1f, bufB, m, hi);
  __syncthreads();
  mlp_layer(bufB, w2T, b2f, bufA, m, hi);
  __syncthreads();
  mlp_layer(bufA, w3T, b3f, bufB, m, hi);
  __syncthreads();

  // ---- layer 4: 128 -> 1 dot + sigmoid; split across half-waves ------------
  {
    const _Float16* hrow = bufB + m * LW + hi * 64;
    const float*    wseg = w4f + hi * 64;
    float s = 0.f;
#pragma unroll 8
    for (int kq = 0; kq < 64; ++kq) s += (float)hrow[kq] * wseg[kq];
    s += __shfl_xor(s, 16, 32);       // combine the two halves
    if (hi == 0) {
      const float t = s + b4f[0];
      out[nodeBase + m] = 1.0f / (1.0f + expf(-t));
    }
  }
}

extern "C" void kernel_launch(void* const* d_in, const int* in_sizes, int n_in,
                              void* d_out, int out_size, void* d_ws, size_t ws_size,
                              hipStream_t stream) {
  (void)n_in; (void)d_ws; (void)ws_size; (void)in_sizes;
  // setup_inputs order: x(0), edge_index(1), edge_attr(2), k(3),
  //                     W1(4), b1(5), W2(6), b2(7), W3(8), b3(9), W4(10), b4(11)
  const float* edge_attr = (const float*)d_in[2];
  const float* W1 = (const float*)d_in[4];
  const float* b1 = (const float*)d_in[5];
  const float* W2 = (const float*)d_in[6];
  const float* b2 = (const float*)d_in[7];
  const float* W3 = (const float*)d_in[8];
  const float* b3 = (const float*)d_in[9];
  const float* W4 = (const float*)d_in[10];
  const float* b4 = (const float*)d_in[11];
  float* out = (float*)d_out;

  const int N = out_size;                        // 200000 nodes
  const int blocks = (N + 16 * WAVES - 1) / (16 * WAVES);  // 8 waves/block, 16 nodes/wave
  gnn_angle_wmma_kernel<<<blocks, 32 * WAVES, SMEM_TOTAL, stream>>>(
      edge_attr, W1, b1, W2, b2, W3, b3, W4, b4, out, N);
}
